// GGRNet_55439437856836
// MI455X (gfx1250) — matrix operand order repaired and checked by reading, not compiled, not run on hardware
//
#include <hip/hip_runtime.h>
#include <hip/hip_bf16.h>
#include <math.h>

typedef __attribute__((ext_vector_type(16))) _Float16 v16h;
typedef __attribute__((ext_vector_type(8)))  _Float16 v8h;
typedef __attribute__((ext_vector_type(8)))  float    v8f;
typedef __attribute__((ext_vector_type(4)))  unsigned int v4u;
typedef __attribute__((ext_vector_type(8)))  int v8i;
typedef __attribute__((ext_vector_type(4)))  int v4i;

#define N_FEAT   128
#define N_GRAPHS 256
#define DIM_H    95
#define N_OUT    12
#define BN_EPS   1e-5f

// packed B-fragment array: 32 fragments (ks 0..3 x nt 0..7) x 32 lanes x 16 halves
#define FRAG_HALVES (32 * 32 * 16)      // 16384 halves = 32 KB

#if defined(__has_builtin)
#if __has_builtin(__builtin_amdgcn_tensor_load_to_lds) && \
    __has_builtin(__builtin_amdgcn_s_wait_tensorcnt)
#define USE_TDM 1
#endif
#endif
#ifndef USE_TDM
#define USE_TDM 0
#endif

// ---------------------------------------------------------------------------
// Pack both 128x128 nn1 weight matrices (f32 row-major, K x N) directly into
// the per-lane WMMA B-fragment layout, in f16:
//   P[((ks*8+nt)*32 + lane)*16 + h] = W[(ks*32 + (lane<16?0:16) + h)*128 + nt*16 + (lane&15)]
// so a wave's B fragment is one contiguous 32-byte read per (ks, nt).
// ---------------------------------------------------------------------------
__global__ void pack_weights(const float* __restrict__ Wa,
                             const float* __restrict__ Wb,
                             _Float16* __restrict__ Pa,
                             _Float16* __restrict__ Pb) {
  int tid = blockIdx.x * blockDim.x + threadIdx.x;
  if (tid >= FRAG_HALVES) return;
  int h    = tid & 15;
  int lane = (tid >> 4) & 31;
  int frag = tid >> 9;             // ks*8 + nt
  int ks = frag >> 3;
  int nt = frag & 7;
  int n = nt * 16 + (lane & 15);
  int k = ks * 32 + ((lane < 16) ? 0 : 16) + h;
  Pa[tid] = (_Float16)Wa[k * N_FEAT + n];
  Pb[tid] = (_Float16)Wb[k * N_FEAT + n];
}

// ---------------------------------------------------------------------------
// Edge aggregation: agg[dst] += x[src]   (segment_sum over edges)
// one thread per (edge, float4-chunk): 32 chunks per 128-feat row
// ---------------------------------------------------------------------------
__global__ void edge_agg(const float* __restrict__ x,
                         const int*   __restrict__ ei,
                         float* __restrict__ agg, int nE) {
  int tid = blockIdx.x * blockDim.x + threadIdx.x;
  int e = tid >> 5;
  int c = tid & 31;
  if (e >= nE) return;
  int src = ei[e];
  int dst = ei[nE + e];
  float4 v = ((const float4*)(x + (size_t)src * N_FEAT))[c];
  float* p = agg + (size_t)dst * N_FEAT + c * 4;
  atomicAdd(p + 0, v.x);
  atomicAdd(p + 1, v.y);
  atomicAdd(p + 2, v.z);
  atomicAdd(p + 3, v.w);
}

// ---------------------------------------------------------------------------
// A16 = f16(x + agg)
// ---------------------------------------------------------------------------
__global__ void prep_a16(const float* __restrict__ x,
                         const float* __restrict__ agg,
                         _Float16* __restrict__ a16, int n) {
  int i = blockIdx.x * blockDim.x + threadIdx.x;
  if (i < n) a16[i] = (_Float16)(x[i] + agg[i]);
}

// ---------------------------------------------------------------------------
// Stage the 32KB packed-fragment array into LDS. Preferred path: Tensor Data
// Mover (tensor_load_to_lds + s_wait_tensorcnt), issued by wave 0 with a flat
// 1-D descriptor: 8192 dword elements, data_size=4B, type=2 ("image").
// ---------------------------------------------------------------------------
__device__ __forceinline__ void stage_frags_to_lds(_Float16* sF,
                                                   const _Float16* __restrict__ P,
                                                   int tid, int nthreads) {
#if USE_TDM
  if (tid < 32) {
    unsigned int ldsa = (unsigned int)(unsigned long long)sF;
    unsigned long long ga = (unsigned long long)P;
    v4u g0;
    g0.x = 1u;                                            // count=1, user desc
    g0.y = ldsa;                                          // lds_addr
    g0.z = (unsigned int)(ga & 0xFFFFFFFFull);            // global_addr[31:0]
    g0.w = (unsigned int)((ga >> 32) & 0x01FFFFFFull)     // global_addr[56:32]
           | (2u << 30);                                  // type=2
    v8i g1;
    g1[0] = (2 << 16);                  // data_size = 4B
    g1[1] = (int)(8192u << 16);         // tensor_dim0[15:0] at bits 63:48
    g1[2] = (1 << 16);                  // tensor_dim0[31:16]=0 ; tensor_dim1=1
    g1[3] = (int)(8192u << 16);         // tensor_dim1 hi=0 ; tile_dim0=8192
    g1[4] = 1;                          // tile_dim1=1, tile_dim2=0
    g1[5] = 8192;                       // tensor_dim0_stride[31:0]
    g1[6] = 0;
    g1[7] = 0;
    v4i gz = {0, 0, 0, 0};
#if __clang_major__ >= 23
    v8i gz8 = {0, 0, 0, 0, 0, 0, 0, 0};
    __builtin_amdgcn_tensor_load_to_lds(g0, g1, gz, gz, gz8, 0);
#else
    __builtin_amdgcn_tensor_load_to_lds(g0, g1, gz, gz, 0);
#endif
    __builtin_amdgcn_s_wait_tensorcnt(0);
  }
#else
  for (int i = tid; i < FRAG_HALVES / 8; i += nthreads)
    ((v8h*)sF)[i] = ((const v8h*)P)[i];
#endif
}

// ---------------------------------------------------------------------------
// WMMA GEMM: out = relu(A[M x 128] @ W[128 x 128] + bias)
//   WRITE_F32 == 0 : write f16 output (GEMM1)
//   WRITE_F32 == 1 : write f32 output + accumulate per-column sum/sumsq (GEMM2)
// Block: 256 threads = 8 waves; wave w owns 16-row strip mt = blockIdx*8 + w,
// loops over all 8 N-tiles. Packed B fragments staged in LDS (TDM when
// available); each fragment is one contiguous v16h LDS read (2x ds_load_b128).
// ---------------------------------------------------------------------------
template <int WRITE_F32>
__global__ void gemm128_wmma(const _Float16* __restrict__ A,
                             const _Float16* __restrict__ P,
                             const float* __restrict__ bias,
                             _Float16* __restrict__ outH,
                             float* __restrict__ outF,
                             float* __restrict__ colsum,
                             float* __restrict__ colsq,
                             int M) {
  __shared__ _Float16 sF[FRAG_HALVES];    // 32 KB packed B fragments
  __shared__ float sBias[N_FEAT];
  __shared__ float sSum[N_FEAT];
  __shared__ float sSq[N_FEAT];

  stage_frags_to_lds(sF, P, threadIdx.x, blockDim.x);
  if (threadIdx.x < N_FEAT) {
    sBias[threadIdx.x] = bias[threadIdx.x];
    sSum[threadIdx.x] = 0.f;
    sSq[threadIdx.x] = 0.f;
  }
  __syncthreads();

  const int wave = threadIdx.x >> 5;
  const int lane = threadIdx.x & 31;
  const int hi16 = (lane < 16) ? 0 : 1;   // upper half-wave flag
  const int l16  = lane & 15;

  const int strips = M >> 4;              // number of 16-row strips
  const int mt = blockIdx.x * 8 + wave;
  const bool active = (mt < strips);      // wave-uniform -> EXEC all-1 inside

  if (active) {
    const int row = mt * 16 + l16;
    // A fragments for k-steps 0..3 (contiguous k order, same as packed B: lanes0-15 K={0..15},
    // lanes16-31 K={16..31} within each 32-wide k-step)
    union AU { v16h v; v8h h[2]; };
    AU afr[4];
#pragma unroll
    for (int ks = 0; ks < 4; ++ks) {
      int k0 = ks * 32 + hi16 * 16;
      afr[ks].h[0] = *(const v8h*)(A + (size_t)row * N_FEAT + k0);
      afr[ks].h[1] = *(const v8h*)(A + (size_t)row * N_FEAT + k0 + 8);
    }

    for (int nt = 0; nt < 8; ++nt) {
      const int n = nt * 16 + l16;
      v8f acc = {};
#pragma unroll
      for (int ks = 0; ks < 4; ++ks) {
        const int fbase = (((ks << 3) | nt) << 5) | lane;   // fragment-lane slot
        v16h bfr = ((const v16h*)sF)[fbase];                // 32B contiguous
        acc = __builtin_amdgcn_wmma_f32_16x16x32_f16(
            false, afr[ks].v, false, bfr, (short)0, acc, false, false);
      }
      const float b = sBias[n];
      float lsum = 0.f, lsq = 0.f;
#pragma unroll
      for (int r = 0; r < 8; ++r) {
        float v = acc[r] + b;
        v = v > 0.f ? v : 0.f;                    // relu epilogue
        const int orow = mt * 16 + hi16 * 8 + r;  // C/D layout row
        if (WRITE_F32) {
          outF[(size_t)orow * N_FEAT + n] = v;
          lsum += v; lsq += v * v;
        } else {
          outH[(size_t)orow * N_FEAT + n] = (_Float16)v;
        }
      }
      if (WRITE_F32) {
        atomicAdd(&sSum[n], lsum);   // ds_add_f32
        atomicAdd(&sSq[n],  lsq);
      }
    }
  }

  if (WRITE_F32) {
    __syncthreads();
    if (threadIdx.x < N_FEAT) {
      atomicAdd(&colsum[threadIdx.x], sSum[threadIdx.x]);
      atomicAdd(&colsq[threadIdx.x],  sSq[threadIdx.x]);
    }
  }
}

// ---------------------------------------------------------------------------
// BN statistics finalize: stats[0..127]=mean, stats[128..255]=rstd
// ---------------------------------------------------------------------------
__global__ void bn_finalize(const float* __restrict__ colsum,
                            const float* __restrict__ colsq,
                            float* __restrict__ stats, float inv_n) {
  int f = threadIdx.x;
  if (f < N_FEAT) {
    float mean = colsum[f] * inv_n;
    float var = colsq[f] * inv_n - mean * mean;
    stats[f] = mean;
    stats[N_FEAT + f] = rsqrtf(var + BN_EPS);
  }
}

// ---------------------------------------------------------------------------
// Graph node counts
// ---------------------------------------------------------------------------
__global__ void count_nodes(const int* __restrict__ batch,
                            float* __restrict__ counts, int n) {
  int i = blockIdx.x * blockDim.x + threadIdx.x;
  if (i < n) atomicAdd(&counts[batch[i]], 1.f);
}

// ---------------------------------------------------------------------------
// BN apply + global pool scatter: pool[batch[i]] += bn(x1[i])
// ---------------------------------------------------------------------------
__global__ void pool_scatter(const float* __restrict__ x1,
                             const int* __restrict__ batch,
                             const float* __restrict__ gamma,
                             const float* __restrict__ beta,
                             const float* __restrict__ stats,
                             float* __restrict__ pool, int nElem) {
  int tid = blockIdx.x * blockDim.x + threadIdx.x;
  if (tid >= nElem) return;
  int i = tid >> 7;
  int f = tid & (N_FEAT - 1);
  float v = gamma[f] * (x1[tid] - stats[f]) * stats[N_FEAT + f] + beta[f];
  atomicAdd(&pool[(size_t)batch[i] * N_FEAT + f], v);
}

// ---------------------------------------------------------------------------
// xg = pool / max(count,1);  h0 = xg
// ---------------------------------------------------------------------------
__global__ void pool_div(const float* __restrict__ pool,
                         const float* __restrict__ counts,
                         float* __restrict__ xg, float* __restrict__ h) {
  int tid = blockIdx.x * blockDim.x + threadIdx.x;
  if (tid >= N_GRAPHS * N_FEAT) return;
  int g = tid >> 7;
  float c = counts[g];
  c = c > 1.f ? c : 1.f;
  float v = pool[tid] / c;
  xg[tid] = v;
  h[tid] = v;
}

// ---------------------------------------------------------------------------
// One gated-recurrence iteration:
//   cat=[xg,h]; p=cat@Wl1+bl1; q=cat@Wl2+bl2; h'=[tanh(q), sigmoid(p)]
// thread (g, j): j<64 -> tanh(q_j), j>=64 -> sigmoid(p_{j-64})
// ---------------------------------------------------------------------------
__global__ void recur_iter(const float* __restrict__ xg,
                           const float* __restrict__ hin,
                           const float* __restrict__ Wl1,
                           const float* __restrict__ bl1,
                           const float* __restrict__ Wl2,
                           const float* __restrict__ bl2,
                           float* __restrict__ hout) {
  int tid = blockIdx.x * blockDim.x + threadIdx.x;
  if (tid >= N_GRAPHS * N_FEAT) return;
  int g = tid >> 7;
  int j = tid & (N_FEAT - 1);
  const float* W;
  float s;
  int o;
  if (j < 64) { W = Wl2; s = bl2[j];      o = j;      }
  else        { W = Wl1; s = bl1[j - 64]; o = j - 64; }
  const float* xgr = xg + (size_t)g * N_FEAT;
  const float* hr  = hin + (size_t)g * N_FEAT;
#pragma unroll 4
  for (int k = 0; k < N_FEAT; ++k) s += xgr[k] * W[k * 64 + o];
#pragma unroll 4
  for (int k = 0; k < N_FEAT; ++k) s += hr[k] * W[(N_FEAT + k) * 64 + o];
  hout[tid] = (j < 64) ? tanhf(s) : (1.f / (1.f + expf(-s)));
}

// ---------------------------------------------------------------------------
// Output MLP
// ---------------------------------------------------------------------------
__global__ void mlp1(const float* __restrict__ h, const float* __restrict__ Wm1,
                     const float* __restrict__ bm1, float* __restrict__ m) {
  int tid = blockIdx.x * blockDim.x + threadIdx.x;
  if (tid >= N_GRAPHS * DIM_H) return;
  int g = tid / DIM_H, o = tid % DIM_H;
  float s = bm1[o];
  const float* hr = h + (size_t)g * N_FEAT;
#pragma unroll 4
  for (int k = 0; k < N_FEAT; ++k) s += hr[k] * Wm1[k * DIM_H + o];
  m[tid] = s > 0.f ? s : 0.f;
}

__global__ void mlp2(const float* __restrict__ m, const float* __restrict__ Wm2,
                     const float* __restrict__ bm2, float* __restrict__ out) {
  int tid = blockIdx.x * blockDim.x + threadIdx.x;
  if (tid >= N_GRAPHS * N_OUT) return;
  int g = tid / N_OUT, o = tid % N_OUT;
  float s = bm2[o];
  const float* mr = m + (size_t)g * DIM_H;
  for (int k = 0; k < DIM_H; ++k) s += mr[k] * Wm2[k * N_OUT + o];
  out[tid] = s;
}

// ---------------------------------------------------------------------------
// Host launcher
// ---------------------------------------------------------------------------
extern "C" void kernel_launch(void* const* d_in, const int* in_sizes, int n_in,
                              void* d_out, int out_size, void* d_ws, size_t ws_size,
                              hipStream_t stream) {
  const float* x     = (const float*)d_in[0];
  const int*   ei    = (const int*)d_in[1];
  const int*   batch = (const int*)d_in[2];
  const float* W1a   = (const float*)d_in[3];
  const float* b1a   = (const float*)d_in[4];
  const float* W1b   = (const float*)d_in[5];
  const float* b1b   = (const float*)d_in[6];
  const float* gamma = (const float*)d_in[7];
  const float* beta  = (const float*)d_in[8];
  const float* Wl1   = (const float*)d_in[9];
  const float* bl1   = (const float*)d_in[10];
  const float* Wl2   = (const float*)d_in[11];
  const float* bl2   = (const float*)d_in[12];
  const float* Wm1   = (const float*)d_in[13];
  const float* bm1   = (const float*)d_in[14];
  const float* Wm2   = (const float*)d_in[15];
  const float* bm2   = (const float*)d_in[16];

  const int nNodes = in_sizes[0] / N_FEAT;   // 50000
  const int nEdges = in_sizes[1] / 2;        // 600000
  const int nElem  = in_sizes[0];            // nodes * 128

  // workspace carve-up (256B aligned)
  char* ws = (char*)d_ws;
  const size_t SZ_NODE_F32 = (size_t)nElem * 4;
  const size_t SZ_NODE_F16 = (size_t)nElem * 2;
  size_t off = 0;
  float*    agg    = (float*)(ws + off);    off += SZ_NODE_F32;
  _Float16* A16    = (_Float16*)(ws + off); off += SZ_NODE_F16;
  _Float16* Pa     = (_Float16*)(ws + off); off += FRAG_HALVES * 2;
  _Float16* Pb     = (_Float16*)(ws + off); off += FRAG_HALVES * 2;
  _Float16* H16    = (_Float16*)(ws + off); off += SZ_NODE_F16;
  float*    x1     = (float*)(ws + off);    off += SZ_NODE_F32;
  float*    colsum = (float*)(ws + off);    off += 512;
  float*    colsq  = (float*)(ws + off);    off += 512;
  float*    stats  = (float*)(ws + off);    off += 1024;
  float*    pool   = (float*)(ws + off);    off += N_GRAPHS * N_FEAT * 4;
  float*    counts = (float*)(ws + off);    off += 1024;
  float*    xg     = (float*)(ws + off);    off += N_GRAPHS * N_FEAT * 4;
  float*    hA     = (float*)(ws + off);    off += N_GRAPHS * N_FEAT * 4;
  float*    hB     = (float*)(ws + off);    off += N_GRAPHS * N_FEAT * 4;
  float*    mbuf   = (float*)(ws + off);    off += ((size_t)N_GRAPHS * DIM_H * 4 + 255) & ~(size_t)255;

  // zero accumulators (fresh every call; capture-safe stream memsets)
  hipMemsetAsync(agg, 0, SZ_NODE_F32, stream);
  hipMemsetAsync(colsum, 0, 512, stream);
  hipMemsetAsync(colsq, 0, 512, stream);
  hipMemsetAsync(pool, 0, N_GRAPHS * N_FEAT * 4, stream);
  hipMemsetAsync(counts, 0, 1024, stream);

  // weight packing into WMMA B-fragment layout
  pack_weights<<<(FRAG_HALVES + 255) / 256, 256, 0, stream>>>(W1a, W1b, Pa, Pb);

  // edge aggregation
  {
    long long t = (long long)nEdges * 32;
    edge_agg<<<(unsigned)((t + 255) / 256), 256, 0, stream>>>(x, ei, agg, nEdges);
  }

  // A16 = f16(x + agg)
  prep_a16<<<(nElem + 255) / 256, 256, 0, stream>>>(x, agg, A16, nElem);

  // GEMM1: H16 = relu(A16 @ W1a + b1a)   [f16 out]
  const int strips = nNodes / 16;
  const int gblocks = (strips + 7) / 8;
  gemm128_wmma<0><<<gblocks, 256, 0, stream>>>(A16, Pa, b1a, H16, nullptr,
                                               nullptr, nullptr, nNodes);
  // GEMM2: x1 = relu(H16 @ W1b + b1b)    [f32 out + column stats]
  gemm128_wmma<1><<<gblocks, 256, 0, stream>>>(H16, Pb, b1b, nullptr, x1,
                                               colsum, colsq, nNodes);

  // BN stats
  bn_finalize<<<1, 128, 0, stream>>>(colsum, colsq, stats, 1.0f / (float)nNodes);

  // pooling
  count_nodes<<<(nNodes + 255) / 256, 256, 0, stream>>>(batch, counts, nNodes);
  pool_scatter<<<(nElem + 255) / 256, 256, 0, stream>>>(x1, batch, gamma, beta,
                                                        stats, pool, nElem);
  pool_div<<<(N_GRAPHS * N_FEAT) / 256, 256, 0, stream>>>(pool, counts, xg, hA);

  // gated recurrence, 10 iterations, ping-pong h buffers (ends in hA)
  float* hin = hA;
  float* hout = hB;
  for (int it = 0; it < 10; ++it) {
    recur_iter<<<(N_GRAPHS * N_FEAT) / 256, 256, 0, stream>>>(xg, hin, Wl1, bl1,
                                                              Wl2, bl2, hout);
    float* t = hin; hin = hout; hout = t;
  }
  // after 10 swaps, final h is in hin (== hA)

  // output MLP
  mlp1<<<(N_GRAPHS * DIM_H + 255) / 256, 256, 0, stream>>>(hin, Wm1, bm1, mbuf);
  mlp2<<<(N_GRAPHS * N_OUT + 255) / 256, 256, 0, stream>>>(mbuf, Wm2, bm2,
                                                           (float*)d_out);
}
